// Selection_34153579938510
// MI455X (gfx1250) — compile-verified, matrix operand order checked by reading, not run on hardware
//
#include <hip/hip_runtime.h>
#include <hip/hip_bf16.h>
#include <math.h>

// ---------------- problem constants (from reference) ----------------
#define BB   16
#define LL   512
#define NN   8192      // B*L
#define CC   512       // dim_enc == dim_z == dim_sprites
#define KK   81        // Km1 + blank
#define KP   96        // K padded to multiple of 32 for WMMA
#define PP   4096      // 4*32*32 sprite pixels
#define WOFF 0                      // weights  (8192*81)
#define LOFF 663552                 // logits_r (512*16*81)
#define POFF 1327104                // log_probs
#define SOFF 1990656                // S        (512*16*4096)

// gfx1250 async global->LDS staging (ASYNCcnt-tracked). Fallback: sync copy.
#ifndef USE_ASYNC_COPY
#define USE_ASYNC_COPY 1
#endif

typedef __attribute__((ext_vector_type(16))) __bf16 v16bf;
typedef __attribute__((ext_vector_type(8)))  float  v8f;

// Build a 16x32 bf16 WMMA fragment for this lane.
// Lane layout (ISA 7.12.2, 16-bit A 16x32): lanes 0-15 hold K {0..7,16..23},
// lanes 16-31 hold K {8..15,24..31}; caller passes koff = (lane>=16)?8:0 and
// a pointer to the start of this lane's row (M for A, N for B-transposed).
__device__ __forceinline__ v16bf frag_bf16(const __bf16* row, int c0, int koff) {
  v16bf f;
  const __bf16* p0 = row + c0 + koff;
  const __bf16* p1 = row + c0 + 16 + koff;
#pragma unroll
  for (int i = 0; i < 8; ++i) { f[i] = p0[i]; f[8 + i] = p1[i]; }
  return f;
}

// ---------------- K1: sprite-latent memory (81x512) + LN + pad ----------------
// grid 96 x 512 threads. Rows 0..79: GEMM row + LayerNorm. Row 80: blank latent
// (no LN, per reference: blank appended AFTER LN). Rows 81..95: zero padding.
__global__ void lat_kernel(const float* __restrict__ latents,
                           const float* __restrict__ Wl,
                           const float* __restrict__ bl,
                           const float* __restrict__ blank,
                           __bf16* __restrict__ lat_bf) {
  const int k = blockIdx.x;
  const int j = threadIdx.x;           // 512 threads == dim
  if (k >= KK) { lat_bf[k * CC + j] = (__bf16)0.0f; return; }
  if (k == KK - 1) { lat_bf[k * CC + j] = (__bf16)blank[j]; return; }

  __shared__ float lrow[CC];
  __shared__ float red[CC];
  __shared__ float stats[2];
  lrow[j] = latents[k * CC + j];
  __syncthreads();

  float acc = bl[j];
  const float* w = Wl + (size_t)j * CC;
  for (int c = 0; c < CC; ++c) acc += lrow[c] * w[c];

  red[j] = acc;
  __syncthreads();
  for (int s = 256; s > 0; s >>= 1) { if (j < s) red[j] += red[j + s]; __syncthreads(); }
  if (j == 0) stats[0] = red[0] * (1.0f / CC);
  __syncthreads();
  const float m = stats[0];
  const float d = acc - m;
  red[j] = d * d;
  __syncthreads();
  for (int s = 256; s > 0; s >>= 1) { if (j < s) red[j] += red[j + s]; __syncthreads(); }
  if (j == 0) stats[1] = rsqrtf(red[0] * (1.0f / CC) + 1e-5f);
  __syncthreads();
  lat_bf[k * CC + j] = (__bf16)(d * stats[1]);
}

// ---------------- K2a: transposed bf16 sprite memory (4096 x 96) ----------------
__global__ void sprite_pack_kernel(const float* __restrict__ prot,
                                   const float* __restrict__ msk,
                                   __bf16* __restrict__ sprite_t) {
  int idx = blockIdx.x * blockDim.x + threadIdx.x;
  const int total = PP * KP;
  for (; idx < total; idx += gridDim.x * blockDim.x) {
    const int p = idx / KP, k = idx % KP;
    float v = 0.0f;
    if (k < KK - 1) {
      const int c = p >> 10, hw = p & 1023;
      v = (c < 3) ? prot[(size_t)k * 3072 + c * 1024 + hw]
                  : msk[(size_t)k * 1024 + hw];
    }
    sprite_t[idx] = (__bf16)v;
  }
}

// ---------------- K2b: W_anchors f32 -> bf16 (one-time pack) ----------------
// Removes per-M-tile re-conversion from the anchors hot loop and halves its
// L2 read traffic (512 blocks each re-read the whole matrix).
__global__ void wa_pack_kernel(const float* __restrict__ Wa,
                               __bf16* __restrict__ wa_bf) {
  int idx = blockIdx.x * blockDim.x + threadIdx.x;
  for (; idx < CC * CC; idx += gridDim.x * blockDim.x)
    wa_bf[idx] = (__bf16)Wa[idx];
}

// ---------------- K3: anchors = LN(xf @ Wa^T + ba), bf16 out ----------------
// One block = 16 rows (one M-tile) x all 512 cols. 256 threads = 8 waves;
// wave w owns N-tiles 4w..4w+3. A staged in LDS (transposed from x's (B,C,L)
// layout), C staged in LDS for the fused LayerNorm.
#define ASTR 520   // padded row stride to spread LDS banks
__global__ void anchors_kernel(const float* __restrict__ x,
                               const __bf16* __restrict__ wa_bf,
                               const float* __restrict__ ba,
                               __bf16* __restrict__ a_bf) {
  __shared__ __bf16 aT[16 * ASTR];
  __shared__ float  cT[16 * ASTR];
  __shared__ float  parts_s[256];
  __shared__ float  parts_q[256];
  __shared__ float  rmean[16];
  __shared__ float  rstd[16];

  const int n0 = blockIdx.x * 16;
  const int b  = n0 >> 9;
  const int l0 = n0 & (LL - 1);
  const int t  = threadIdx.x;

  // stage A tile: aT[l][c] = x[b][c][l0+l], converted to bf16
  const float* xb = x + (size_t)b * CC * LL;
#pragma unroll
  for (int it = 0; it < 32; ++it) {
    const int i = t + it * 256;
    const int c = i >> 4, ll = i & 15;
    aT[ll * ASTR + c] = (__bf16)xb[(size_t)c * LL + l0 + ll];
  }
  __syncthreads();

  const int wv = t >> 5, lane = t & 31;
  const int ln = lane & 15, koff = (lane >> 4) << 3;
  const int mbase = (lane >> 4) << 3;

  v8f acc[4];
#pragma unroll
  for (int j = 0; j < 4; ++j)
#pragma unroll
    for (int e = 0; e < 8; ++e) acc[j][e] = 0.0f;

  const __bf16* arow = aT + ln * ASTR;
  for (int c0 = 0; c0 < CC; c0 += 32) {
    const v16bf af = frag_bf16(arow, c0, koff);
#pragma unroll
    for (int j = 0; j < 4; ++j) {
      const int col = (wv * 4 + j) * 16 + ln;
      const v16bf bf = frag_bf16(wa_bf + (size_t)col * CC, c0, koff);
      acc[j] = __builtin_amdgcn_wmma_f32_16x16x32_bf16(
          false, af, false, bf, (short)0, acc[j], false, false);
    }
  }

  // bias + spill C to LDS (C/D layout: lane = N col, vgpr r = M (+8 hi half))
#pragma unroll
  for (int j = 0; j < 4; ++j) {
    const int col = (wv * 4 + j) * 16 + ln;
    const float bb = ba[col];
#pragma unroll
    for (int r = 0; r < 8; ++r)
      cT[(mbase + r) * ASTR + col] = acc[j][r] + bb;
  }
  __syncthreads();

  // fused LayerNorm over each of the 16 rows
  {
    const int row = t >> 4, cidx = t & 15;
    float s = 0.0f, q = 0.0f;
#pragma unroll
    for (int i = 0; i < 32; ++i) {
      const float v = cT[row * ASTR + cidx + 16 * i];
      s += v; q += v * v;
    }
    parts_s[t] = s; parts_q[t] = q;
  }
  __syncthreads();
  if (t < 16) {
    float s = 0.0f, q = 0.0f;
    for (int i = 0; i < 16; ++i) { s += parts_s[t * 16 + i]; q += parts_q[t * 16 + i]; }
    const float m = s * (1.0f / CC);
    rmean[t] = m;
    rstd[t]  = rsqrtf(q * (1.0f / CC) - m * m + 1e-5f);
  }
  __syncthreads();
#pragma unroll
  for (int it = 0; it < 32; ++it) {
    const int i = t + it * 256;
    const int row = i >> 9, col = i & (CC - 1);
    const float v = (cT[row * ASTR + col] - rmean[row]) * rstd[row];
    a_bf[(size_t)(n0 + row) * CC + col] = (__bf16)v;
  }
}

// ---------------- K4: logits + softmax + log_softmax + bf16 weights ----------------
// One block = 16 rows x 96 cols (6 WMMA N-tiles on 6 waves, 192 threads), K=512.
__global__ void logits_kernel(const __bf16* __restrict__ a_bf,
                              const __bf16* __restrict__ lat_bf,
                              float* __restrict__ out,
                              __bf16* __restrict__ w_bf) {
  __shared__ float lg[16][KP];
  const int n0 = blockIdx.x * 16;
  const int t = threadIdx.x, wv = t >> 5, lane = t & 31;
  const int ln = lane & 15, koff = (lane >> 4) << 3;
  const int mbase = (lane >> 4) << 3;

  {
    v8f acc;
#pragma unroll
    for (int e = 0; e < 8; ++e) acc[e] = 0.0f;
    const __bf16* arow = a_bf  + (size_t)(n0 + ln) * CC;
    const __bf16* brow = lat_bf + (size_t)(wv * 16 + ln) * CC;
    for (int c0 = 0; c0 < CC; c0 += 32) {
      const v16bf af = frag_bf16(arow, c0, koff);
      const v16bf bf = frag_bf16(brow, c0, koff);
      acc = __builtin_amdgcn_wmma_f32_16x16x32_bf16(
          false, af, false, bf, (short)0, acc, false, false);
    }
    const float inv = 0.044194173824159216f;  // 1/sqrt(512)
#pragma unroll
    for (int r = 0; r < 8; ++r)
      lg[mbase + r][wv * 16 + ln] = acc[r] * inv;
  }
  __syncthreads();

  if (t < 16) {
    const int row = t, n = n0 + row;
    const int bb = n >> 9, l = n & (LL - 1);
    float mx = -1e30f;
    for (int k = 0; k < KK; ++k) mx = fmaxf(mx, lg[row][k]);
    float se = 0.0f;
    for (int k = 0; k < KK; ++k) se += __expf(lg[row][k] - mx);
    const float lse = __logf(se), rse = 1.0f / se;
    float* wout = out + WOFF + (size_t)n * KK;
    float* lro  = out + LOFF + ((size_t)l * BB + bb) * KK;
    float* lpo  = out + POFF + ((size_t)l * BB + bb) * KK;
    __bf16* wb  = w_bf + (size_t)n * KP;
    for (int k = 0; k < KK; ++k) {
      const float v = lg[row][k];
      const float w = __expf(v - mx) * rse;
      wout[k] = w;
      lro[k]  = v;
      lpo[k]  = v - mx - lse;
      wb[k]   = (__bf16)w;
    }
    for (int k = KK; k < KP; ++k) wb[k] = (__bf16)0.0f;
  }
}

// ---------------- K5: S = weights(8192x96) @ sprite_t^T -> (L,B,4,32,32) ----------------
// One block = 64 rows (4 M-tiles) x one 512-col chunk. The 512x96 bf16 sprite
// chunk (96 KB contiguous) is staged in LDS once via async global->LDS b128
// copies (ASYNCcnt), then reused by all 4 M-tiles: 4x less L2 sprite traffic.
// LDS rows padded to 104 bf16 (52-bank stride) so the 16 fragment rows per
// wave land on distinct bank groups for ds_load_b128.
#define SSTR 104
__global__ void sgemm_kernel(const __bf16* __restrict__ w_bf,
                             const __bf16* __restrict__ sprite_t,
                             float* __restrict__ out) {
  __shared__ __bf16 spr[512 * SSTR];   // 106,496 B

  const int blk = blockIdx.x;          // 1024 = 128 row-groups * 8 chunks
  const int mg = blk >> 3, ch = blk & 7;
  const int t = threadIdx.x, wv = t >> 5, lane = t & 31;
  const int ln = lane & 15, koff = (lane >> 4) << 3;
  const int mbase = (lane >> 4) << 3;

  // ---- stage sprite chunk: 512 rows x 192B, 12 x b128 per row ----
  {
    const char* gsrc = (const char*)(sprite_t + (size_t)ch * 512 * KP);
#if USE_ASYNC_COPY
    const unsigned lbase = (unsigned)(size_t)(&spr[0]);
    for (int i = t; i < 512 * 12; i += 256) {
      const int row = i / 12, part = i - row * 12;
      const unsigned long long ga =
          (unsigned long long)(gsrc + (size_t)row * 192 + part * 16);
      const unsigned la = lbase + (unsigned)(row * (SSTR * 2) + part * 16);
      asm volatile("global_load_async_to_lds_b128 %0, %1, off"
                   :: "v"(la), "v"(ga) : "memory");
    }
    asm volatile("s_wait_asynccnt 0x0" ::: "memory");
#else
    for (int i = t; i < 512 * 12; i += 256) {
      const int row = i / 12, part = i - row * 12;
      const uint4 v = *(const uint4*)(gsrc + (size_t)row * 192 + part * 16);
      *(uint4*)((char*)&spr[row * SSTR] + part * 16) = v;
    }
#endif
  }
  __syncthreads();

  float* Sout = out + SOFF;
#pragma unroll
  for (int mt = 0; mt < 4; ++mt) {
    const int n0 = mg * 64 + mt * 16;
    const __bf16* arow = w_bf + (size_t)(n0 + ln) * KP;
    v16bf af[3];
#pragma unroll
    for (int s = 0; s < 3; ++s) af[s] = frag_bf16(arow, s * 32, koff);

#pragma unroll
    for (int j = 0; j < 4; ++j) {
      const __bf16* brow = spr + ((wv * 4 + j) * 16 + ln) * SSTR;
      v8f acc;
#pragma unroll
      for (int e = 0; e < 8; ++e) acc[e] = 0.0f;
#pragma unroll
      for (int s = 0; s < 3; ++s) {
        const v16bf bf = frag_bf16(brow, s * 32, koff);
        acc = __builtin_amdgcn_wmma_f32_16x16x32_bf16(
            false, af[s], false, bf, (short)0, acc, false, false);
      }
      const int pcol = ch * 512 + (wv * 4 + j) * 16 + ln;
#pragma unroll
      for (int r = 0; r < 8; ++r) {
        const int n = n0 + mbase + r;
        const int bb = n >> 9, l = n & (LL - 1);
        Sout[(size_t)l * (BB * PP) + (size_t)bb * PP + pcol] = acc[r];
      }
    }
  }
}

// ---------------- launcher ----------------
extern "C" void kernel_launch(void* const* d_in, const int* in_sizes, int n_in,
                              void* d_out, int out_size, void* d_ws, size_t ws_size,
                              hipStream_t stream) {
  (void)in_sizes; (void)n_in; (void)out_size; (void)ws_size;
  const float* x       = (const float*)d_in[0];
  const float* latents = (const float*)d_in[1];
  const float* Wl      = (const float*)d_in[2];
  const float* bl      = (const float*)d_in[3];
  const float* Wa      = (const float*)d_in[4];
  const float* ba      = (const float*)d_in[5];
  const float* blank   = (const float*)d_in[6];
  const float* prot    = (const float*)d_in[7];
  const float* msk     = (const float*)d_in[8];
  float* out = (float*)d_out;

  char* ws = (char*)d_ws;
  __bf16* lat_bf   = (__bf16*)(ws);                       // 96*512
  __bf16* a_bf     = (__bf16*)(ws + 98304);               // 8192*512
  __bf16* w_bf     = (__bf16*)(ws + 8486912);             // 8192*96
  __bf16* sprite_t = (__bf16*)(ws + 10059776);            // 4096*96
  __bf16* wa_bf    = (__bf16*)(ws + 10846208);            // 512*512

  lat_kernel<<<KP, CC, 0, stream>>>(latents, Wl, bl, blank, lat_bf);
  sprite_pack_kernel<<<384, 256, 0, stream>>>(prot, msk, sprite_t);
  wa_pack_kernel<<<256, 256, 0, stream>>>(Wa, wa_bf);
  anchors_kernel<<<NN / 16, 256, 0, stream>>>(x, wa_bf, ba, a_bf);
  logits_kernel<<<NN / 16, 192, 0, stream>>>(a_bf, lat_bf, out, w_bf);
  sgemm_kernel<<<128 * 8, 256, 0, stream>>>(w_bf, sprite_t, out);
}
